// MoEDecoderLayer_34411277975756
// MI455X (gfx1250) — matrix operand order repaired and checked by reading, not compiled, not run on hardware
//
#include <hip/hip_runtime.h>
#include <hip/hip_bf16.h>
#include <math.h>
#include <stdint.h>

// ---------- types ----------
typedef __attribute__((ext_vector_type(16))) __bf16 v16bf;
typedef __attribute__((ext_vector_type(8)))  __bf16 v8bf;
typedef __attribute__((ext_vector_type(8)))  float  v8f;
typedef __attribute__((ext_vector_type(4)))  int    v4i_t;

// D = A(16x32 bf16) * B(32x16 bf16) + C(16x16 f32)
__device__ __forceinline__ v8f wmma_bf16(v16bf a, v16bf b, v8f c) {
  return __builtin_amdgcn_wmma_f32_16x16x32_bf16(false, a, false, b, (short)0, c,
                                                 false, false);
}

// A fragment: 16x32 bf16 tile, row-major in LDS with row stride ldk (elems).
// lanes 0-15: rows 0..15, K=0..7 (VGPR0-3) and K=16..23 (VGPR4-7);
// lanes 16-31: K=8..15 and K=24..31.
__device__ __forceinline__ v16bf load_frag_a(const __bf16* sm, int ldk) {
  int lane = threadIdx.x & 31;
  int row  = lane & 15;
  int hf   = lane >> 4;
  const __bf16* p = sm + row * ldk + hf * 8;
  v16bf r;
  ((v8bf*)&r)[0] = *(const v8bf*)(p);
  ((v8bf*)&r)[1] = *(const v8bf*)(p + 16);
  return r;
}

// B fragment: 32x16 bf16 (KxN). LDS tile stored [n][k] row-major, stride ldk.
// lanes 0-15: col N=lane, K=0..7 and K=16..23; lanes 16-31: K=8..15 and K=24..31.
__device__ __forceinline__ v16bf load_frag_b(const __bf16* sm, int ldk) {
  int lane = threadIdx.x & 31;
  int col  = lane & 15;
  int hf   = lane >> 4;
  const __bf16* p = sm + col * ldk + hf * 8;
  v16bf r;
  ((v8bf*)&r)[0] = *(const v8bf*)(p);
  ((v8bf*)&r)[1] = *(const v8bf*)(p + 16);
  return r;
}

// ---------- CDNA5 async global->LDS copy (16B per lane), ASYNCcnt tracked ----
typedef __attribute__((address_space(1))) v4i_t* gv4_t;  // global int4*
typedef __attribute__((address_space(3))) v4i_t* lv4_t;  // LDS int4*

__device__ __forceinline__ void async_cp16(const void* g, void* l) {
#if __has_builtin(__builtin_amdgcn_global_load_async_to_lds_b128)
  __builtin_amdgcn_global_load_async_to_lds_b128(
      (gv4_t)(unsigned long long)(uintptr_t)g,
      (lv4_t)(unsigned)(uintptr_t)l, 0, 0);
#else
  asm volatile("global_load_async_to_lds_b128 %0, %1, off"
               :: "v"((unsigned)(uintptr_t)l),
                  "v"((unsigned long long)(uintptr_t)g)
               : "memory");
#endif
}

__device__ __forceinline__ void wait_asynccnt0() {
#if __has_builtin(__builtin_amdgcn_s_wait_asynccnt)
  __builtin_amdgcn_s_wait_asynccnt(0);
#else
  asm volatile("s_wait_asynccnt 0x0" ::: "memory");
#endif
}

// =====================================================================
// Generic WMMA GEMM:  out = act @ W^T + bias   (W is [N x K] row-major)
// 128x128 block tile, 256 threads (8 waves = 2 row x 4 col), K-step 32.
// MODE 0: dense rows (T, N, K multiples of tile -> branch-free)
// MODE 1: gathered MoE fc1: rows from list[e] (aid>>1 = token), out = H[base[e]+g], +ReLU
// MODE 2: MoE fc2: act rows = H[base[e]+g], out row = aid (Y2)
// =====================================================================
#define GTM 128
#define GTN 128
#define GTK 32

template <int MODE, int RELU>
__global__ __launch_bounds__(256) void gemm_kernel(
    const float* __restrict__ act, const float* __restrict__ wmat,
    const float* __restrict__ bias, float* __restrict__ out,
    int T, int K, int N, int rowTilesPerE,
    size_t wstride, int bstride,
    const int* __restrict__ list, const int* __restrict__ cnt,
    const int* __restrict__ basep, int Tcap) {
  __shared__ __bf16 a_sm[GTM * GTK];
  __shared__ __bf16 b_sm[GTN * GTK];

  const int tid = threadIdx.x;
  const int e   = blockIdx.x / rowTilesPerE;
  const int rt  = blockIdx.x % rowTilesPerE;
  const int colbase = blockIdx.y * GTN;
  const int rowbase = rt * GTM;

  int nrows = T;
  if (MODE != 0) {
    nrows = cnt[e];
    if (rowbase >= nrows) return;  // uniform across block
  }
  const float* W  = wmat + (size_t)e * wstride;
  const float* Bp = bias + (size_t)e * bstride;

  // Precompute this thread's 4 source-row pointers for A staging (row = idx>>3).
  const float* arow[4];
  int avalid[4];
#pragma unroll
  for (int it = 0; it < 4; it++) {
    int idx = tid + it * 256;
    int r   = idx >> 3;
    if (MODE == 0) {
      arow[it] = act + (size_t)(rowbase + r) * K;
      avalid[it] = 1;
    } else if (MODE == 1) {
      int g = rowbase + r;
      avalid[it] = (g < nrows);
      int aid = avalid[it] ? list[e * Tcap + g] : 0;
      arow[it] = act + (size_t)(aid >> 1) * K;
    } else {
      int g = rowbase + r;
      avalid[it] = (g < nrows);
      arow[it] = act + (size_t)(basep[e] + (avalid[it] ? g : 0)) * K;
    }
  }

  v8f acc[4][2];
#pragma unroll
  for (int i = 0; i < 4; i++)
#pragma unroll
    for (int j = 0; j < 2; j++)
#pragma unroll
      for (int q = 0; q < 8; q++) acc[i][j][q] = 0.f;

  const int wave = tid >> 5;
  const int wr   = wave & 1;   // 2 row-waves of 64 rows
  const int wc   = wave >> 1;  // 4 col-waves of 32 cols

  for (int k0 = 0; k0 < K; k0 += GTK) {
    // ---- stage A tile (128 x 32) fp32 -> bf16 ----
#pragma unroll
    for (int it = 0; it < 4; it++) {
      int idx = tid + it * 256;
      int r   = idx >> 3;
      int c4  = (idx & 7) << 2;
      float4 v = make_float4(0.f, 0.f, 0.f, 0.f);
      if (MODE == 0 || avalid[it])
        v = *(const float4*)(arow[it] + k0 + c4);
      __bf16* d = a_sm + r * GTK + c4;
      d[0] = (__bf16)v.x; d[1] = (__bf16)v.y; d[2] = (__bf16)v.z; d[3] = (__bf16)v.w;
    }
    // ---- stage B tile (128 n x 32 k) from W[N x K] ----
#pragma unroll
    for (int it = 0; it < 4; it++) {
      int idx = tid + it * 256;
      int r   = idx >> 3;
      int c4  = (idx & 7) << 2;
      float4 v = *(const float4*)(W + (size_t)(colbase + r) * K + k0 + c4);
      __bf16* d = b_sm + r * GTK + c4;
      d[0] = (__bf16)v.x; d[1] = (__bf16)v.y; d[2] = (__bf16)v.z; d[3] = (__bf16)v.w;
    }
    if (k0 + GTK < K)  // touch next K-slab of W (global_prefetch)
      __builtin_prefetch(W + (size_t)(colbase + (tid >> 1)) * K + k0 + GTK);
    __syncthreads();

    v16bf af[4], bfr[2];
#pragma unroll
    for (int i = 0; i < 4; i++)
      af[i] = load_frag_a(a_sm + (wr * 64 + i * 16) * GTK, GTK);
#pragma unroll
    for (int j = 0; j < 2; j++)
      bfr[j] = load_frag_b(b_sm + (wc * 32 + j * 16) * GTK, GTK);
#pragma unroll
    for (int i = 0; i < 4; i++)
#pragma unroll
      for (int j = 0; j < 2; j++)
        acc[i][j] = wmma_bf16(af[i], bfr[j], acc[i][j]);
    __syncthreads();
  }

  // ---- epilogue ----
  const int lane = tid & 31;
  const int hf   = lane >> 4;
  const int lcol = lane & 15;
#pragma unroll
  for (int i = 0; i < 4; i++) {
#pragma unroll
    for (int j = 0; j < 2; j++) {
      int gcol = colbase + wc * 32 + j * 16 + lcol;
      float bv = Bp[gcol];
#pragma unroll
      for (int q = 0; q < 8; q++) {
        int lrow = wr * 64 + i * 16 + q + 8 * hf;
        float v = acc[i][j][q] + bv;
        if (RELU) v = fmaxf(v, 0.f);
        if (MODE == 0) {
          out[(size_t)(rowbase + lrow) * N + gcol] = v;
        } else {
          int g = rowbase + lrow;
          if (g < nrows) {
            if (MODE == 1) {
              out[(size_t)(basep[e] + g) * N + gcol] = v;
            } else {
              int aid = list[e * Tcap + g];
              out[(size_t)aid * N + gcol] = v;
            }
          }
        }
      }
    }
  }
}

// =====================================================================
// Flash attention: 128 threads = 4 waves, 64 queries/block, HD=64.
// Q/K staged via CDNA5 async global->LDS; V staged transposed.
// Q/K/V bf16 [BH, Slen, 64]; out fp32 into [B*Sq, D] at col h*64.
// =====================================================================
__global__ __launch_bounds__(128) void flash_kernel(
    const __bf16* __restrict__ Qb, const __bf16* __restrict__ Kb,
    const __bf16* __restrict__ Vb, float* __restrict__ out,
    int Sq, int Skv, int Hn, int Dm, float scale) {
  __shared__ __bf16 q_sm[64 * 64];
  __shared__ __bf16 k_sm[32 * 64];
  __shared__ __bf16 vt_sm[64 * 32];  // [hd][key]
  __shared__ __bf16 p_sm[4][16 * 32];

  const int bh = blockIdx.x;
  const int b  = bh / Hn;
  const int h  = bh % Hn;
  const int q0 = blockIdx.y * 64;
  const int tid = threadIdx.x;
  const int wave = tid >> 5;
  const int lane = tid & 31;
  const int hf   = lane >> 4;
  const int lcol = lane & 15;

  // ---- async-stage Q (64x64 bf16 = 8KB, 512 x 16B chunks) ----
  const char* Qp = (const char*)(Qb + ((size_t)bh * Sq + q0) * 64);
#pragma unroll
  for (int it = 0; it < 4; it++) {
    int idx = tid + it * 128;
    async_cp16(Qp + idx * 16, (char*)q_sm + idx * 16);
  }
  wait_asynccnt0();
  __syncthreads();

  v16bf qa0 = load_frag_a(q_sm + wave * 16 * 64, 64);
  v16bf qa1 = load_frag_a(q_sm + wave * 16 * 64 + 32, 64);

  float m_i[8], l_i[8];
  v8f o[4];
#pragma unroll
  for (int q = 0; q < 8; q++) { m_i[q] = -3.0e38f; l_i[q] = 0.f; }
#pragma unroll
  for (int j = 0; j < 4; j++)
#pragma unroll
    for (int q = 0; q < 8; q++) o[j][q] = 0.f;

  for (int kt = 0; kt < Skv; kt += 32) {
    const char* Kp = (const char*)(Kb + ((size_t)bh * Skv + kt) * 64);
    const __bf16* Vp = Vb + ((size_t)bh * Skv + kt) * 64;
    // async-stage K (32x64 bf16 = 4KB, 256 x 16B chunks)
#pragma unroll
    for (int it = 0; it < 2; it++) {
      int idx = tid + it * 128;
      async_cp16(Kp + idx * 16, (char*)k_sm + idx * 16);
    }
    // stage V transposed [hd][key]
#pragma unroll
    for (int it = 0; it < 16; it++) {
      int idx = tid + it * 128;
      int kk = idx >> 6, hd = idx & 63;
      vt_sm[hd * 32 + kk] = Vp[idx];
    }
    wait_asynccnt0();
    __syncthreads();

    // S = Q @ K^T (two 16-key groups)
    v8f s[2];
#pragma unroll
    for (int g = 0; g < 2; g++) {
      v8f z;
#pragma unroll
      for (int q = 0; q < 8; q++) z[q] = 0.f;
      v16bf kb0 = load_frag_b(k_sm + g * 16 * 64, 64);       // hd 0..31
      v16bf kb1 = load_frag_b(k_sm + g * 16 * 64 + 32, 64);  // hd 32..63
      z = wmma_bf16(qa0, kb0, z);
      z = wmma_bf16(qa1, kb1, z);
#pragma unroll
      for (int q = 0; q < 8; q++) z[q] *= scale;
      s[g] = z;
    }

    // per-row (16 lanes/half) online softmax update
    float mt[8];
#pragma unroll
    for (int q = 0; q < 8; q++) mt[q] = fmaxf(s[0][q], s[1][q]);
#pragma unroll
    for (int off = 1; off < 16; off <<= 1)
#pragma unroll
      for (int q = 0; q < 8; q++) mt[q] = fmaxf(mt[q], __shfl_xor(mt[q], off, 32));

    float pexp[2][8], rs[8], alpha[8];
#pragma unroll
    for (int q = 0; q < 8; q++) {
      float nm = fmaxf(m_i[q], mt[q]);
      alpha[q] = __expf(m_i[q] - nm);
      pexp[0][q] = __expf(s[0][q] - nm);
      pexp[1][q] = __expf(s[1][q] - nm);
      rs[q] = pexp[0][q] + pexp[1][q];
      m_i[q] = nm;
    }
#pragma unroll
    for (int off = 1; off < 16; off <<= 1)
#pragma unroll
      for (int q = 0; q < 8; q++) rs[q] += __shfl_xor(rs[q], off, 32);
#pragma unroll
    for (int q = 0; q < 8; q++) l_i[q] = l_i[q] * alpha[q] + rs[q];
#pragma unroll
    for (int j = 0; j < 4; j++)
#pragma unroll
      for (int q = 0; q < 8; q++) o[j][q] *= alpha[q];

    // C-layout P -> A-layout via LDS
    __bf16* pw = p_sm[wave];
#pragma unroll
    for (int g = 0; g < 2; g++)
#pragma unroll
      for (int q = 0; q < 8; q++)
        pw[(q + 8 * hf) * 32 + lcol + 16 * g] = (__bf16)pexp[g][q];
    __syncthreads();

    v16bf pa = load_frag_a(pw, 32);
#pragma unroll
    for (int j = 0; j < 4; j++)
      o[j] = wmma_bf16(pa, load_frag_b(vt_sm + j * 16 * 32, 32), o[j]);
    __syncthreads();
  }

  // epilogue: normalize, write to [T, D] at col h*64
#pragma unroll
  for (int q = 0; q < 8; q++) {
    float inv = 1.f / l_i[q];
    int t = b * Sq + q0 + wave * 16 + q + 8 * hf;
#pragma unroll
    for (int j = 0; j < 4; j++)
      out[(size_t)t * Dm + h * 64 + j * 16 + lcol] = o[j][q] * inv;
  }
}

// =====================================================================
// Helpers
// =====================================================================
__global__ void repack_kernel(const float* __restrict__ src, __bf16* __restrict__ dst,
                              int Slen, int C, int co, int Hn, int Bn) {
  size_t idx = (size_t)blockIdx.x * 256 + threadIdx.x;
  size_t total = (size_t)Bn * Hn * Slen * 64;
  if (idx >= total) return;
  int hd = idx & 63;
  size_t r = idx >> 6;
  int s = r % Slen; r /= Slen;
  int h = r % Hn;
  int b = r / Hn;
  dst[idx] = (__bf16)src[(size_t)(b * Slen + s) * C + co + h * 64 + hd];
}

__global__ __launch_bounds__(256) void ln_kernel(
    const float* __restrict__ a, const float* __restrict__ bq,
    const float* __restrict__ g, const float* __restrict__ be,
    float* __restrict__ out, int D) {
  __shared__ float red[256];
  const int row = blockIdx.x;
  const int tid = threadIdx.x;
  const float* ap = a + (size_t)row * D;
  const float* bp = bq + (size_t)row * D;
  float s = 0.f;
  for (int d = tid; d < D; d += 256) s += ap[d] + bp[d];
  red[tid] = s; __syncthreads();
  for (int off = 128; off > 0; off >>= 1) {
    if (tid < off) red[tid] += red[tid + off];
    __syncthreads();
  }
  float mu = red[0] / D;
  __syncthreads();
  float s2 = 0.f;
  for (int d = tid; d < D; d += 256) {
    float v = ap[d] + bp[d] - mu;
    s2 += v * v;
  }
  red[tid] = s2; __syncthreads();
  for (int off = 128; off > 0; off >>= 1) {
    if (tid < off) red[tid] += red[tid + off];
    __syncthreads();
  }
  float rstd = rsqrtf(red[0] / D + 1e-5f);
  for (int d = tid; d < D; d += 256)
    out[(size_t)row * D + d] = (ap[d] + bp[d] - mu) * rstd * g[d] + be[d];
}

__global__ __launch_bounds__(256) void router_kernel(
    const float* __restrict__ x, const float* __restrict__ wg,
    const float* __restrict__ bg, const float* __restrict__ wn,
    const float* __restrict__ bn, const float* __restrict__ eps,
    float* __restrict__ noisy, int D, int E) {
  __shared__ float xs[1024];
  __shared__ float red[256];
  const int t = blockIdx.x;
  const int tid = threadIdx.x;
  for (int d = tid; d < D; d += 256) xs[d] = x[(size_t)t * D + d];
  __syncthreads();
  const int e = tid >> 4, seg = tid & 15;
  const int c0 = seg * (D / 16);
  float pg = 0.f, pn = 0.f;
  for (int d = c0; d < c0 + D / 16; d++) {
    pg += xs[d] * wg[e * D + d];
    pn += xs[d] * wn[e * D + d];
  }
  red[tid] = pg; __syncthreads();
  for (int off = 8; off > 0; off >>= 1) {
    if (seg < off) red[tid] += red[tid + off];
    __syncthreads();
  }
  float gsum = red[e * 16];
  __syncthreads();
  red[tid] = pn; __syncthreads();
  for (int off = 8; off > 0; off >>= 1) {
    if (seg < off) red[tid] += red[tid + off];
    __syncthreads();
  }
  float nsum = red[e * 16];
  if (seg == 0) {
    float nl = nsum + bn[e];
    float sp = (nl > 20.f) ? nl : log1pf(__expf(nl));
    noisy[(size_t)t * E + e] = gsum + bg[e] + eps[(size_t)t * E + e] * sp;
  }
}

__global__ void zero_kernel(int* p, int n) {
  int i = blockIdx.x * blockDim.x + threadIdx.x;
  if (i < n) p[i] = 0;
}

__global__ void topk_kernel(const float* __restrict__ noisy, int* cnt,
                            int* __restrict__ list, float* __restrict__ gates,
                            int T, int E, int Tcap) {
  int t = blockIdx.x * blockDim.x + threadIdx.x;
  if (t >= T) return;
  const float* r = noisy + (size_t)t * E;
  float v1 = -3.0e38f, v2 = -3.0e38f;
  int i1 = 0, i2 = 0;
  for (int e = 0; e < E; e++) {
    float v = r[e];
    if (v > v1) { v2 = v1; i2 = i1; v1 = v; i1 = e; }
    else if (v > v2) { v2 = v; i2 = e; }
  }
  float g2 = __expf(v2 - v1);
  float den = 1.f + g2;
  gates[t * 2 + 0] = 1.f / den;
  gates[t * 2 + 1] = g2 / den;
  int p1 = atomicAdd(&cnt[i1], 1);
  list[i1 * Tcap + p1] = t * 2 + 0;
  int p2 = atomicAdd(&cnt[i2], 1);
  list[i2 * Tcap + p2] = t * 2 + 1;
}

__global__ void scan_kernel(const int* cnt, int* basep, int E) {
  if (threadIdx.x == 0 && blockIdx.x == 0) {
    int acc = 0;
    for (int e = 0; e < E; e++) { basep[e] = acc; acc += cnt[e]; }
  }
}

__global__ void combine_kernel(const float* __restrict__ Y2,
                               const float* __restrict__ gates,
                               float* __restrict__ moe, int T, int D) {
  size_t i = (size_t)blockIdx.x * 256 + threadIdx.x;
  if (i >= (size_t)T * D) return;
  int t = i / D;
  int d = i % D;
  moe[i] = gates[t * 2 + 0] * Y2[(size_t)(t * 2 + 0) * D + d] +
           gates[t * 2 + 1] * Y2[(size_t)(t * 2 + 1) * D + d];
}

// =====================================================================
extern "C" void kernel_launch(void* const* d_in, const int* in_sizes, int n_in,
                              void* d_out, int out_size, void* d_ws, size_t ws_size,
                              hipStream_t stream) {
  const int Bn = 4, Sn = 1024, Mn = 1024, Dn = 1024, Hn = 16, En = 16, Fn = 2048;
  const int T = Bn * Sn;  // 4096 (== B*M as well)

  const float* tgt      = (const float*)d_in[0];
  const float* memory   = (const float*)d_in[1];
  const float* eps      = (const float*)d_in[2];
  const float* sa_in_w  = (const float*)d_in[3];
  const float* sa_in_b  = (const float*)d_in[4];
  const float* sa_out_w = (const float*)d_in[5];
  const float* sa_out_b = (const float*)d_in[6];
  const float* ma_in_w  = (const float*)d_in[7];
  const float* ma_in_b  = (const float*)d_in[8];
  const float* ma_out_w = (const float*)d_in[9];
  const float* ma_out_b = (const float*)d_in[10];
  const float* wg = (const float*)d_in[11];
  const float* bg = (const float*)d_in[12];
  const float* wn = (const float*)d_in[13];
  const float* bn = (const float*)d_in[14];
  const float* w1 = (const float*)d_in[15];
  const float* b1 = (const float*)d_in[16];
  const float* w2 = (const float*)d_in[17];
  const float* b2 = (const float*)d_in[18];
  const float* ln1g = (const float*)d_in[19];
  const float* ln1b = (const float*)d_in[20];
  const float* ln2g = (const float*)d_in[21];
  const float* ln2b = (const float*)d_in[22];
  const float* ln3g = (const float*)d_in[23];
  const float* ln3b = (const float*)d_in[24];

  char* wsp = (char*)d_ws;
  auto alloc = [&](size_t bytes) -> void* {
    void* p = (void*)wsp;
    wsp += (bytes + 255) & ~(size_t)255;
    return p;
  };
  float*  y0    = (float*)alloc((size_t)T * 3 * Dn * 4);
  __bf16* qb    = (__bf16*)alloc((size_t)T * Dn * 2);
  __bf16* kb    = (__bf16*)alloc((size_t)T * Dn * 2);
  __bf16* vb    = (__bf16*)alloc((size_t)T * Dn * 2);
  float*  attn  = (float*)alloc((size_t)T * Dn * 4);
  float*  x1    = (float*)alloc((size_t)T * Dn * 4);
  float*  x2    = (float*)alloc((size_t)T * Dn * 4);
  float*  noisy = (float*)alloc((size_t)T * En * 4);
  float*  gates = (float*)alloc((size_t)T * 2 * 4);
  int*    cnt   = (int*)alloc(64);
  int*    basep = (int*)alloc(64);
  int*    list  = (int*)alloc((size_t)En * T * 4);
  float*  Hbuf  = (float*)alloc((size_t)2 * T * Fn * 4);
  float*  Y2    = (float*)alloc((size_t)2 * T * Dn * 4);
  float*  moe   = (float*)alloc((size_t)T * Dn * 4);
  float*  xout  = (float*)d_out;

  const int rpt = T / GTM;  // dense row tiles (32)
  zero_kernel<<<1, 32, 0, stream>>>(cnt, En);

  // ---------------- self attention ----------------
  {
    dim3 g(rpt, 3 * Dn / GTN);
    gemm_kernel<0, 0><<<g, 256, 0, stream>>>(tgt, sa_in_w, sa_in_b, y0, T, Dn,
                                             3 * Dn, rpt, 0, 0,
                                             nullptr, nullptr, nullptr, 0);
  }
  {
    int blocks = (int)(((size_t)T * Dn + 255) / 256);
    repack_kernel<<<blocks, 256, 0, stream>>>(y0, qb, Sn, 3 * Dn, 0,      Hn, Bn);
    repack_kernel<<<blocks, 256, 0, stream>>>(y0, kb, Sn, 3 * Dn, Dn,     Hn, Bn);
    repack_kernel<<<blocks, 256, 0, stream>>>(y0, vb, Sn, 3 * Dn, 2 * Dn, Hn, Bn);
  }
  {
    dim3 g(Bn * Hn, Sn / 64);
    flash_kernel<<<g, 128, 0, stream>>>(qb, kb, vb, attn, Sn, Sn, Hn, Dn, 0.125f);
  }
  {
    dim3 g(rpt, Dn / GTN);
    gemm_kernel<0, 0><<<g, 256, 0, stream>>>(attn, sa_out_w, sa_out_b, y0, T, Dn,
                                             Dn, rpt, 0, 0,
                                             nullptr, nullptr, nullptr, 0);
  }
  ln_kernel<<<T, 256, 0, stream>>>(tgt, y0, ln1g, ln1b, x1, Dn);

  // ---------------- cross attention ----------------
  {
    dim3 gq(rpt, Dn / GTN);
    gemm_kernel<0, 0><<<gq, 256, 0, stream>>>(x1, ma_in_w, ma_in_b, y0, T, Dn, Dn,
                                              rpt, 0, 0, nullptr, nullptr, nullptr, 0);
    dim3 gkv(rpt, 2 * Dn / GTN);  // B*M = 4096 rows
    gemm_kernel<0, 0><<<gkv, 256, 0, stream>>>(memory, ma_in_w + (size_t)Dn * Dn,
                                               ma_in_b + Dn, y0 + (size_t)T * Dn,
                                               Bn * Mn, Dn, 2 * Dn, rpt, 0, 0,
                                               nullptr, nullptr, nullptr, 0);
  }
  {
    int blocks = (int)(((size_t)T * Dn + 255) / 256);
    repack_kernel<<<blocks, 256, 0, stream>>>(y0, qb, Sn, Dn, 0, Hn, Bn);
    repack_kernel<<<blocks, 256, 0, stream>>>(y0 + (size_t)T * Dn, kb, Mn, 2 * Dn, 0,  Hn, Bn);
    repack_kernel<<<blocks, 256, 0, stream>>>(y0 + (size_t)T * Dn, vb, Mn, 2 * Dn, Dn, Hn, Bn);
  }
  {
    dim3 g(Bn * Hn, Sn / 64);
    flash_kernel<<<g, 128, 0, stream>>>(qb, kb, vb, attn, Sn, Mn, Hn, Dn, 0.125f);
  }
  {
    dim3 g(rpt, Dn / GTN);
    gemm_kernel<0, 0><<<g, 256, 0, stream>>>(attn, ma_out_w, ma_out_b, y0, T, Dn,
                                             Dn, rpt, 0, 0,
                                             nullptr, nullptr, nullptr, 0);
  }
  ln_kernel<<<T, 256, 0, stream>>>(x1, y0, ln2g, ln2b, x2, Dn);

  // ---------------- MoE: router + sparse top-2 dispatch ----------------
  router_kernel<<<T, 256, 0, stream>>>(x2, wg, bg, wn, bn, eps, noisy, Dn, En);
  topk_kernel<<<T / 256, 256, 0, stream>>>(noisy, cnt, list, gates, T, En, T);
  scan_kernel<<<1, 32, 0, stream>>>(cnt, basep, En);
  {
    dim3 g1(En * rpt, Fn / GTN);  // fc1 + relu -> H (gathered rows)
    gemm_kernel<1, 1><<<g1, 256, 0, stream>>>(x2, w1, b1, Hbuf, T, Dn, Fn, rpt,
                                              (size_t)Fn * Dn, Fn,
                                              list, cnt, basep, T);
    dim3 g2(En * rpt, Dn / GTN);  // fc2 -> per-assignment Y2
    gemm_kernel<2, 0><<<g2, 256, 0, stream>>>(Hbuf, w2, b2, Y2, T, Fn, Dn, rpt,
                                              (size_t)Dn * Fn, Dn,
                                              list, cnt, basep, T);
  }
  {
    int blocks = (int)(((size_t)T * Dn + 255) / 256);
    combine_kernel<<<blocks, 256, 0, stream>>>(Y2, gates, moe, T, Dn);
  }
  ln_kernel<<<T, 256, 0, stream>>>(x2, moe, ln3g, ln3b, xout, Dn);
}
